// AFFN_11252814315713
// MI455X (gfx1250) — matrix operand order repaired and checked by reading, not compiled, not run on hardware
//
#include <hip/hip_runtime.h>

// ---------------------------------------------------------------------------
// AFFN forward, collapsed to 3 dense bf16 WMMA GEMMs + log-softmax reduce.
// L=64, H=64, IN=4, B=4096.  Weights are pre-masked block-triangular, so the
// scan is exactly:  A0=elu(Xoh@W0+b0); A1=elu(A0@W1+b1); out=elu(A1@W2+b2);
// logp[b] = sum_j log_softmax(out[b,j,:])[x[b,j]].
// GEMM staging: gfx1250 async global->LDS (ASYNCcnt), double-buffered so the
// DMA overlaps the v_wmma stream.
// ---------------------------------------------------------------------------

typedef __attribute__((ext_vector_type(16))) __bf16 v16bf;
typedef __attribute__((ext_vector_type(8)))  float  v8f;
typedef __attribute__((ext_vector_type(4)))  int    v4i;

#define Bsz 4096
#define Lsz 64

// ---- gfx1250 async global->LDS staging (ASYNCcnt path), guarded -----------
__device__ __forceinline__ void stage16(const __bf16* g, __bf16* l) {
#if __has_builtin(__builtin_amdgcn_global_load_async_to_lds_b128)
  __builtin_amdgcn_global_load_async_to_lds_b128(
      (__attribute__((address_space(1))) v4i*)(unsigned long long)g,
      (__attribute__((address_space(3))) v4i*)l,
      /*offset=*/0, /*cpol=*/0);
#else
  *reinterpret_cast<uint4*>(l) = *reinterpret_cast<const uint4*>(g);
#endif
}

__device__ __forceinline__ void stage_wait() {
#if __has_builtin(__builtin_amdgcn_global_load_async_to_lds_b128) && \
    __has_builtin(__builtin_amdgcn_s_wait_asynccnt)
  __builtin_amdgcn_s_wait_asynccnt(0);
#endif
}

// ---------------- prep: f32 (KxN) -> bf16 transposed (NxK) ------------------
__global__ __launch_bounds__(256)
void transpose_to_bf16(const float* __restrict__ W, __bf16* __restrict__ Wt,
                       int K, int N) {
  __shared__ float tile[32][33];
  const int k0 = blockIdx.y * 32;
  const int n0 = blockIdx.x * 32;
  const int tx = threadIdx.x, ty = threadIdx.y;      // 32 x 8
  #pragma unroll
  for (int i = 0; i < 32; i += 8)
    tile[ty + i][tx] = W[(size_t)(k0 + ty + i) * N + (n0 + tx)];
  __syncthreads();
  #pragma unroll
  for (int i = 0; i < 32; i += 8)
    Wt[(size_t)(n0 + ty + i) * K + (k0 + tx)] = (__bf16)tile[tx][ty + i];
}

// ---------------- prep: shifted one-hot input (B x 256) bf16 ----------------
__global__ __launch_bounds__(256)
void build_onehot(const int* __restrict__ x, __bf16* __restrict__ Xoh) {
  const int t = blockIdx.x * blockDim.x + threadIdx.x;   // b*64 + site
  const int b = t >> 6, site = t & 63;
  const int xv = (site >= 1) ? x[b * Lsz + site - 1] : -1;
  const size_t base = (size_t)b * 256 + site * 4;
  #pragma unroll
  for (int i = 0; i < 4; ++i)
    Xoh[base + i] = (__bf16)((i == xv) ? 1.0f : 0.0f);
}

// ---------------- GEMM: C = elu(A(MxK) @ Bt(NxK)^T + bias[N]) ---------------
// 256 threads = 8 waves; block tile 128x128; wave tile 64(M) x 32(N);
// double-buffered BK=32 stages: async DMA for stage s+1 overlaps the
// 8 v_wmma_f32_16x16x32_bf16 per wave of stage s.
template <bool OUT_BF16>
__global__ __launch_bounds__(256)
void gemm_bias_elu(const __bf16* __restrict__ A,    // M x K row-major
                   const __bf16* __restrict__ Bt,   // N x K row-major (B^T)
                   const float*  __restrict__ bias, // N
                   void* __restrict__ Out,          // M x N row-major
                   int M, int N, int K) {
  // padded stride 40 bf16 = 80 B: 16B-aligned rows, conflict-free uint4 reads
  __shared__ __bf16 As[2][128][40];
  __shared__ __bf16 Bs[2][128][40];

  const int tid  = threadIdx.x;
  const int lane = tid & 31;
  const int wave = tid >> 5;
  const int wm   = wave >> 2;            // 0..1 -> 64 rows each
  const int wn   = wave & 3;             // 0..3 -> 32 cols each
  const int l15  = lane & 15;
  const int half = lane >> 4;            // 0 or 1
  const int m_block = blockIdx.y * 128;
  const int n_block = blockIdx.x * 128;

  v8f acc[4][2];
  #pragma unroll
  for (int i = 0; i < 4; ++i)
    #pragma unroll
    for (int j = 0; j < 2; ++j)
      #pragma unroll
      for (int e = 0; e < 8; ++e) acc[i][j][e] = 0.0f;

  // ---- async global -> LDS: 2 passes x 256 threads x 16 B per matrix ----
  auto stage_tile = [&](int k0, int buf) {
    #pragma unroll
    for (int p = 0; p < 2; ++p) {
      const int lin = p * 256 + tid;       // 0..511
      const int row = lin >> 2;            // 0..127
      const int c8  = (lin & 3) << 3;      // 0,8,16,24
      stage16(&A [(size_t)(m_block + row) * K + k0 + c8], &As[buf][row][c8]);
      stage16(&Bt[(size_t)(n_block + row) * K + k0 + c8], &Bs[buf][row][c8]);
    }
  };

  // ---- 8 WMMAs on the resident buffer ----
  auto compute_tile = [&](int buf) {
    // A 16x32: lane<16 row M=l15 K=0..15; lane>=16 K=16..31
    v16bf afrag[4], bfrag[2];
    #pragma unroll
    for (int mt = 0; mt < 4; ++mt) {
      const int r  = wm * 64 + mt * 16 + l15;
      const int kb = half * 16;
      uint4* av = reinterpret_cast<uint4*>(&afrag[mt]);
      av[0] = *reinterpret_cast<const uint4*>(&As[buf][r][kb]);
      av[1] = *reinterpret_cast<const uint4*>(&As[buf][r][kb + 8]);
    }
    // B 32x16: lane<16 col N=l15 K=0..15; lane>=16 K=16..31 (Bs is N-major)
    #pragma unroll
    for (int nt = 0; nt < 2; ++nt) {
      const int c  = wn * 32 + nt * 16 + l15;
      const int kb = half * 16;
      uint4* bv = reinterpret_cast<uint4*>(&bfrag[nt]);
      bv[0] = *reinterpret_cast<const uint4*>(&Bs[buf][c][kb]);
      bv[1] = *reinterpret_cast<const uint4*>(&Bs[buf][c][kb + 8]);
    }
    #pragma unroll
    for (int mt = 0; mt < 4; ++mt)
      #pragma unroll
      for (int nt = 0; nt < 2; ++nt)
        acc[mt][nt] = __builtin_amdgcn_wmma_f32_16x16x32_bf16(
            false, afrag[mt], false, bfrag[nt],
            (short)0, acc[mt][nt], false, false);
  };

  const int S = K >> 5;                  // number of 32-deep stages
  stage_tile(0, 0);
  stage_wait();
  __syncthreads();

  for (int s = 0; s < S - 1; ++s) {
    const int buf = s & 1;
    stage_tile((s + 1) << 5, buf ^ 1);   // async DMA overlaps the WMMAs below
    compute_tile(buf);
    stage_wait();                        // next tile landed (this wave)
    __syncthreads();                     // ... and all waves; readers of buf done
  }
  compute_tile((S - 1) & 1);

  // ---- epilogue: bias + ELU, store per C/D layout (VGPR r -> row r|r+8) ----
  #pragma unroll
  for (int mt = 0; mt < 4; ++mt) {
    const int mbase = m_block + wm * 64 + mt * 16 + half * 8;
    #pragma unroll
    for (int nt = 0; nt < 2; ++nt) {
      const int ncol = n_block + wn * 32 + nt * 16 + l15;
      const float bn = bias[ncol];
      #pragma unroll
      for (int r = 0; r < 8; ++r) {
        float v = acc[mt][nt][r] + bn;
        v = (v > 0.0f) ? v : (__expf(v) - 1.0f);           // ELU
        const size_t off = (size_t)(mbase + r) * N + ncol;
        if (OUT_BF16) reinterpret_cast<__bf16*>(Out)[off] = (__bf16)v;
        else          reinterpret_cast<float*>(Out)[off]  = v;
      }
    }
  }
}

// ---------------- logp[b] = sum_j log_softmax(out[b,j,:])[x[b,j]] ----------
__global__ __launch_bounds__(256)
void logp_kernel(const float* __restrict__ OutElu,   // B x 256 (already elu'd)
                 const int* __restrict__ x,
                 float* __restrict__ logp) {
  const int b = blockIdx.x * blockDim.x + threadIdx.x;
  if (b >= Bsz) return;
  const float* row = OutElu + (size_t)b * 256;
  float acc = 0.0f;
  #pragma unroll 4
  for (int j = 0; j < Lsz; ++j) {
    const float v0 = row[j * 4 + 0], v1 = row[j * 4 + 1];
    const float v2 = row[j * 4 + 2], v3 = row[j * 4 + 3];
    const float m  = fmaxf(fmaxf(v0, v1), fmaxf(v2, v3));
    const float s  = __expf(v0 - m) + __expf(v1 - m) +
                     __expf(v2 - m) + __expf(v3 - m);
    const float lse = m + __logf(s);
    acc += row[j * 4 + x[b * Lsz + j]] - lse;
  }
  logp[b] = acc;
}

// ---------------------------------------------------------------------------
extern "C" void kernel_launch(void* const* d_in, const int* in_sizes, int n_in,
                              void* d_out, int out_size, void* d_ws, size_t ws_size,
                              hipStream_t stream) {
  const int*   x  = (const int*)  d_in[0];
  const float* W0 = (const float*)d_in[1];  // (64,4,64,64)  == (256 , 4096) row-major
  const float* W1 = (const float*)d_in[2];  // (64,64,64,64) == (4096, 4096) row-major
  const float* W2 = (const float*)d_in[3];  // (64,64,64,4)  == (4096, 256 ) row-major
  const float* b0 = (const float*)d_in[4];  // 4096
  const float* b1 = (const float*)d_in[5];  // 4096
  const float* b2 = (const float*)d_in[6];  // 256

  char* ws = (char*)d_ws;
  __bf16* Xoh = (__bf16*)(ws + ((size_t)0   << 20));  //  2 MB: 4096 x 256
  __bf16* W0t = (__bf16*)(ws + ((size_t)2   << 20));  //  2 MB: 4096 x 256
  __bf16* W1t = (__bf16*)(ws + ((size_t)4   << 20));  // 32 MB: 4096 x 4096
  __bf16* W2t = (__bf16*)(ws + ((size_t)36  << 20));  //  2 MB: 256  x 4096
  __bf16* A0  = (__bf16*)(ws + ((size_t)38  << 20));  // 32 MB: 4096 x 4096
  __bf16* A1  = (__bf16*)(ws + ((size_t)70  << 20));  // 32 MB: 4096 x 4096
  float*  H2  = (float*) (ws + ((size_t)102 << 20));  //  4 MB: 4096 x 256

  // weight transposes (f32 -> bf16, N-major for clean WMMA-B loads)
  transpose_to_bf16<<<dim3(4096/32, 256/32),  dim3(32, 8), 0, stream>>>(W0, W0t, 256,  4096);
  transpose_to_bf16<<<dim3(4096/32, 4096/32), dim3(32, 8), 0, stream>>>(W1, W1t, 4096, 4096);
  transpose_to_bf16<<<dim3(256/32,  4096/32), dim3(32, 8), 0, stream>>>(W2, W2t, 4096, 256);
  build_onehot<<<(Bsz * Lsz) / 256, 256, 0, stream>>>(x, Xoh);

  // layer 1: A0 = elu(Xoh @ W0 + b0)        (M=4096, N=4096, K=256)
  gemm_bias_elu<true ><<<dim3(4096/128, 4096/128), 256, 0, stream>>>(
      Xoh, W0t, b0, (void*)A0, Bsz, 4096, 256);
  // layer 2: A1 = elu(A0 @ W1 + b1)         (M=4096, N=4096, K=4096)
  gemm_bias_elu<true ><<<dim3(4096/128, 4096/128), 256, 0, stream>>>(
      A0, W1t, b1, (void*)A1, Bsz, 4096, 4096);
  // layer 3: H2 = elu(A1 @ W2 + b2)         (M=4096, N=256,  K=4096)
  gemm_bias_elu<false><<<dim3(256/128, 4096/128), 256, 0, stream>>>(
      A1, W2t, b2, (void*)H2, Bsz, 256, 4096);

  // final log-softmax gather/reduce
  logp_kernel<<<Bsz / 256, 256, 0, stream>>>(H2, x, (float*)d_out);
}
